// TCRAggregator_87290915324566
// MI455X (gfx1250) — compile-verified, matrix operand order checked by reading, not run on hardware
//
#include <hip/hip_runtime.h>

// Problem constants (match the JAX reference).
#define B_N 200000
#define D_N 128
#define L_N 256
#define H_N 64

#define ACC_STRIDE 144      // 128 feature cols + col 128 = Z, padded to 9*16
#define MSF 36              // LDS row stride (floats) for f32 mask tile (16B-aligned rows)
#define GS 40               // LDS row stride (ushort) for G tile
#define AGG_STEPS 25        // 25 * 32 = 800 batch rows per aggregation block

typedef __attribute__((ext_vector_type(16))) __bf16 v16bf;
typedef __attribute__((ext_vector_type(8)))  float  v8f;

union ABPack { unsigned u[8]; v16bf v; };

// ---- f32 -> packed bf16x2 via V_PERM_B32 (no bf16-cvt builtin on this toolchain)
// exact truncation (for 0.0/1.0 mask values): 1 VALU per pair
__device__ __forceinline__ unsigned pk_exact01(unsigned lo_bits, unsigned hi_bits) {
  return __builtin_amdgcn_perm(hi_bits, lo_bits, 0x07060302u);
}
// round-to-nearest (ties-away) for general values: 3 VALU per pair
__device__ __forceinline__ unsigned pk_rn(float lo, float hi) {
  unsigned a = __float_as_uint(lo) + 0x8000u;
  unsigned b = __float_as_uint(hi) + 0x8000u;
  return __builtin_amdgcn_perm(b, a, 0x07060302u);
}
__device__ __forceinline__ unsigned short bf_one(float x) {
  return (unsigned short)(pk_rn(x, x) & 0xFFFFu);
}

// K index held by VGPR r / lane-half, per CDNA5 16-bit A/B matrix striping
// (ISA 7.12.2: VGPR0..3 hold K=0..7 (lanes 0-15) / K=8..15 (lanes 16-31),
//  VGPR4..7 hold K=16..23 / K=24..31; two bf16 per dword).
__device__ __forceinline__ int kbase(int r, int khalf) {
  return ((r & 4) ? 16 : 0) + 8 * khalf + 2 * (r & 3);
}

// order-preserving float <-> uint key for atomicMax on signed floats
__device__ __forceinline__ unsigned fkey(float f) {
  unsigned u = __float_as_uint(f);
  return (u & 0x80000000u) ? ~u : (u | 0x80000000u);
}
__device__ __forceinline__ float keyf(unsigned k) {
  return __uint_as_float((k & 0x80000000u) ? (k & 0x7FFFFFFFu) : ~k);
}

// ---- CDNA5 async global->LDS copy (ASYNCcnt-tracked, no VGPR round trip)
__device__ __forceinline__ void async_ld_b128(unsigned lds_off, const float* g) {
  asm volatile("global_load_async_to_lds_b128 %0, %1, off"
               :: "v"(lds_off), "v"(g) : "memory");
}
__device__ __forceinline__ void wait_async0() {
#if __has_builtin(__builtin_amdgcn_s_wait_asynccnt)
  __builtin_amdgcn_s_wait_asynccnt(0);
#else
  asm volatile("s_wait_asynccnt 0x0" ::: "memory");
#endif
}

// ---------------------------------------------------------------- init
__global__ void init_ws(float* __restrict__ accbuf, unsigned* __restrict__ maxcell) {
  int i = blockIdx.x * blockDim.x + threadIdx.x;
  if (i < L_N * ACC_STRIDE) accbuf[i] = 0.0f;
  if (i == 0) *maxcell = 0u;   // decodes below any real float key
}

// ------------------------------------------------- logits (bf16 WMMA MLP)
// 256 threads = 8 waves; each wave computes logits for 16 batch rows.
__launch_bounds__(256)
__global__ void logits_kernel(const float* __restrict__ feat,
                              const float* __restrict__ W1,
                              const float* __restrict__ b1,
                              const float* __restrict__ W2,
                              const float* __restrict__ b2,
                              float* __restrict__ logits,
                              unsigned* __restrict__ maxcell) {
  __shared__ __align__(16) unsigned short w1t[H_N * 136]; // W1^T bf16, [h][k]
  const int tid = threadIdx.x;
  // stage W1^T as bf16 k-pairs: one pack + one b32 store per pair
  for (int p = tid; p < (D_N / 2) * H_N; p += 256) {
    int k2 = (p >> 6) * 2, h = p & 63;
    *(unsigned*)&w1t[h * 136 + k2] = pk_rn(W1[k2 * H_N + h], W1[(k2 + 1) * H_N + h]);
  }
  __syncthreads();

  const int wave = tid >> 5, lane = tid & 31;
  const int n16 = lane & 15, khalf = lane >> 4;
  const long row0 = ((long)blockIdx.x * 8 + wave) * 16;

  v8f acc[4];
  #pragma unroll
  for (int nt = 0; nt < 4; ++nt) { v8f z = {}; acc[nt] = z; }

  long rowA = row0 + n16; if (rowA >= B_N) rowA = B_N - 1; // clamp for loads only
  const float* fr = feat + rowA * D_N;

  #pragma unroll
  for (int ks = 0; ks < 4; ++ks) {                         // K = 128 in steps of 32
    ABPack a;
    #pragma unroll
    for (int r = 0; r < 8; ++r) {
      int k = ks * 32 + kbase(r, khalf);
      float2 f2 = *(const float2*)(fr + k);
      a.u[r] = pk_rn(f2.x, f2.y);
    }
    const unsigned short* wr = &w1t[n16 * 136 + ks * 32];
    ABPack bm[2];
    #pragma unroll
    for (int r = 0; r < 8; ++r)                            // preload tile nt=0
      bm[0].u[r] = *(const unsigned*)(wr + kbase(r, khalf));
    #pragma unroll
    for (int nt = 0; nt < 4; ++nt) {                       // H = 64 -> 4 N-tiles
      if (nt < 3) {                                        // ping-pong next tile
        #pragma unroll
        for (int r = 0; r < 8; ++r)
          bm[(nt + 1) & 1].u[r] =
              *(const unsigned*)(wr + (nt + 1) * 16 * 136 + kbase(r, khalf));
      }
      acc[nt] = __builtin_amdgcn_wmma_f32_16x16x32_bf16(
          false, a.v, false, bm[nt & 1].v, (short)0, acc[nt], false, false);
    }
  }

  float b1v[4], w2v[4];
  #pragma unroll
  for (int nt = 0; nt < 4; ++nt) {
    b1v[nt] = b1[nt * 16 + n16];
    w2v[nt] = W2[nt * 16 + n16];
  }
  float s[8];
  #pragma unroll
  for (int r = 0; r < 8; ++r) {
    float t = 0.0f;
    #pragma unroll
    for (int nt = 0; nt < 4; ++nt)
      t += tanhf(acc[nt][r] + b1v[nt]) * w2v[nt];
    #pragma unroll
    for (int off = 1; off < 16; off <<= 1)                 // reduce 16-lane halves
      t += __shfl_xor(t, off, 32);
    s[r] = t;
  }
  if (n16 == 0) {                                          // lanes 0 and 16 emit rows
    float lmax = -3.0e38f; bool any = false;
    const float bias2 = b2[0];
    #pragma unroll
    for (int r = 0; r < 8; ++r) {
      long row = row0 + khalf * 8 + r;                     // C layout: M = r + 8*khalf
      if (row < B_N) {
        float lg = s[r] + bias2;
        logits[row] = lg;
        lmax = fmaxf(lmax, lg); any = true;
      }
    }
    if (any) atomicMax(maxcell, fkey(lmax));
  }
}

// ------------------------------------------------------------- e = exp(l - M)
__global__ void exp_kernel(float* __restrict__ logits,
                           const unsigned* __restrict__ maxcell) {
  int i = blockIdx.x * blockDim.x + threadIdx.x;
  if (i < B_N) {
    float M = keyf(*maxcell);
    logits[i] = expf(logits[i] - M);
  }
}

// --------------------------------------- split-K aggregation (bf16 WMMA GEMM)
// 512 threads = 16 waves; wave w owns library tile w (16 rows of L=256).
// Per 32-wide K step:
//   - mask tile [256 x 32] f32 streamed global->LDS with async-to-LDS b128
//     (overlaps the VALU G staging; exact perm-pack to bf16 at gather time)
//   - Gt[d][b] = bf16(e_b * F[b,d]) staged via perm-pack pairs; row 128 = e_b
//     so Z arrives as GEMM column 128 of a 9th N-tile
//   - 9 x v_wmma_f32_16x16x32_bf16 per wave, B-tile ping-ponged in registers
__launch_bounds__(512)
__global__ void aggregate_kernel(const float* __restrict__ mask,
                                 const float* __restrict__ feat,
                                 const float* __restrict__ evals,
                                 float* __restrict__ accbuf) {
  __shared__ __align__(16) float MskF[L_N * MSF];          // 36.9 KB f32 mask tile
  __shared__ __align__(16) unsigned short Gt[144 * GS];    // 11.5 KB
  const int tid = threadIdx.x;
  const long base = (long)blockIdx.x * (AGG_STEPS * 32);
  if (base >= B_N) return;

  for (int i = tid; i < 15 * GS; i += 512) Gt[129 * GS + i] = 0; // pad rows 129..143

  const int wave = tid >> 5, lane = tid & 31;
  const int n16 = lane & 15, khalf = lane >> 4;

  v8f acc[9];
  #pragma unroll
  for (int nt = 0; nt < 9; ++nt) { v8f z = {}; acc[nt] = z; }

  for (int step = 0; step < AGG_STEPS; ++step) {
    const long b0 = base + (long)step * 32;
    __syncthreads();                                       // prev compute done

    { // ---- issue async mask stream: [256 x 32] f32 -> LDS (4 x b128 / thread)
      const int l = tid >> 1, half = tid & 1;
      const float* mrow = mask + (long)l * B_N + b0 + half * 16;
      const unsigned ldsrow =
          (unsigned)(size_t)&MskF[l * MSF + half * 16];    // low 32b = LDS offset
      if (b0 + 32 <= B_N) {
        #pragma unroll
        for (int j = 0; j < 4; ++j)
          async_ld_b128(ldsrow + j * 16u, mrow + j * 4);
        __builtin_prefetch(mrow + 32, 0, 1);               // global_prefetch_b8
      } else {
        for (int j = 0; j < 16; ++j) {
          long b = b0 + half * 16 + j;
          MskF[l * MSF + half * 16 + j] = (b < B_N) ? mrow[j] : 0.0f;
        }
      }
    }
    { // ---- stage Gt: thread -> 2 consecutive b cols x 4 d rows (packed stores)
      const int dg = tid & 31, bp = tid >> 5;              // dg: d-group, bp: b-pair
      const int d0 = dg * 4, bl = bp * 2;
      const long b = b0 + bl;
      float ev0 = 0.0f, ev1 = 0.0f;
      float4 f0 = {0, 0, 0, 0}, f1 = {0, 0, 0, 0};
      if (b < B_N)     { ev0 = evals[b];     f0 = *(const float4*)(feat + b * D_N + d0); }
      if (b + 1 < B_N) { ev1 = evals[b + 1]; f1 = *(const float4*)(feat + (b + 1) * D_N + d0); }
      unsigned* g = (unsigned*)&Gt[d0 * GS + bl];
      g[0 * GS / 2] = pk_rn(ev0 * f0.x, ev1 * f1.x);
      g[1 * GS / 2] = pk_rn(ev0 * f0.y, ev1 * f1.y);
      g[2 * GS / 2] = pk_rn(ev0 * f0.z, ev1 * f1.z);
      g[3 * GS / 2] = pk_rn(ev0 * f0.w, ev1 * f1.w);
      if (dg == 0) *(unsigned*)&Gt[128 * GS + bl] = pk_rn(ev0, ev1);
    }
    wait_async0();                                         // s_wait_asynccnt 0
    __syncthreads();                                       // all staging visible

    ABPack a;                                              // A = mask (exact perm pack)
    #pragma unroll
    for (int r = 0; r < 8; ++r) {
      float2 m2 = *(const float2*)&MskF[(wave * 16 + n16) * MSF + kbase(r, khalf)];
      a.u[r] = pk_exact01(__float_as_uint(m2.x), __float_as_uint(m2.y));
    }
    const unsigned short* gr = &Gt[n16 * GS];
    ABPack bm[2];
    #pragma unroll
    for (int r = 0; r < 8; ++r)                            // preload tile nt=0
      bm[0].u[r] = *(const unsigned*)(gr + kbase(r, khalf));
    #pragma unroll
    for (int nt = 0; nt < 9; ++nt) {
      if (nt < 8) {                                        // ping-pong next tile
        #pragma unroll
        for (int r = 0; r < 8; ++r)
          bm[(nt + 1) & 1].u[r] =
              *(const unsigned*)(gr + (nt + 1) * 16 * GS + kbase(r, khalf));
      }
      acc[nt] = __builtin_amdgcn_wmma_f32_16x16x32_bf16(
          false, a.v, false, bm[nt & 1].v, (short)0, acc[nt], false, false);
    }
  }

  // flush split-K partials (f32 accumulate in L2)
  #pragma unroll
  for (int nt = 0; nt < 9; ++nt) {
    int c = nt * 16 + n16;
    if (c < 129) {
      #pragma unroll
      for (int r = 0; r < 8; ++r) {
        int l = wave * 16 + r + 8 * khalf;                 // C layout: M = r + 8*khalf
        atomicAdd(&accbuf[l * ACC_STRIDE + c], acc[nt][r]);
      }
    }
  }
}

// ---------------------------------------------------------------- finalize
__global__ void finalize_kernel(const float* __restrict__ accbuf,
                                float* __restrict__ out) {
  int i = blockIdx.x * blockDim.x + threadIdx.x;
  if (i < L_N * D_N) {
    int l = i >> 7, d = i & 127;
    float Z = accbuf[l * ACC_STRIDE + 128];
    out[i] = (Z > 0.0f) ? accbuf[l * ACC_STRIDE + d] / Z : 0.0f;
  }
}

// ---------------------------------------------------------------- launch
extern "C" void kernel_launch(void* const* d_in, const int* in_sizes, int n_in,
                              void* d_out, int out_size, void* d_ws, size_t ws_size,
                              hipStream_t stream) {
  const float* feat = (const float*)d_in[0];
  const float* mask = (const float*)d_in[1];
  const float* W1   = (const float*)d_in[2];
  const float* b1   = (const float*)d_in[3];
  const float* W2   = (const float*)d_in[4];
  const float* b2   = (const float*)d_in[5];
  float* out = (float*)d_out;

  // workspace: [accbuf L*144 f32][logits/e B f32][maxcell u32]  (~950 KB)
  float* accbuf = (float*)d_ws;
  float* logits = accbuf + (size_t)L_N * ACC_STRIDE;
  unsigned* maxcell = (unsigned*)(logits + B_N);

  init_ws<<<(L_N * ACC_STRIDE + 255) / 256, 256, 0, stream>>>(accbuf, maxcell);
  logits_kernel<<<(B_N + 127) / 128, 256, 0, stream>>>(feat, W1, b1, W2, b2,
                                                       logits, maxcell);
  exp_kernel<<<(B_N + 255) / 256, 256, 0, stream>>>(logits, maxcell);
  aggregate_kernel<<<(B_N + AGG_STEPS * 32 - 1) / (AGG_STEPS * 32), 512, 0,
                     stream>>>(mask, feat, logits, accbuf);
  finalize_kernel<<<(L_N * D_N + 255) / 256, 256, 0, stream>>>(accbuf, out);
}
